// GraphCNN_87729001988408
// MI455X (gfx1250) — matrix operand order repaired and checked by reading, not compiled, not run on hardware
//
#include <hip/hip_runtime.h>
#include <stdint.h>

typedef __attribute__((ext_vector_type(16))) __bf16 v16bf;
typedef __attribute__((ext_vector_type(8)))  float  v8f;

union Frag { v16bf v; uint4 q[2]; };

#define HID 256

// ---------------- hardware f32 atomic add (CDNA5 global_atomic_add_f32) ----
__device__ __forceinline__ void gatomic_add(float* p, float v) {
    asm volatile("global_atomic_add_f32 %0, %1, off"
                 :: "v"((unsigned long long)(uintptr_t)p), "v"(v)
                 : "memory");
}

// ---------------- small utility kernels ------------------------------------
__global__ void k_zero(float* p, int n) {
    int i = blockIdx.x * blockDim.x + threadIdx.x;
    if (i < n) p[i] = 0.0f;
}

__global__ void k_copy4(const float4* __restrict__ src, float4* __restrict__ dst, int n4) {
    int i = blockIdx.x * blockDim.x + threadIdx.x;
    if (i < n4) dst[i] = src[i];
}

__global__ void k_init_score(float* out, const float* __restrict__ b0,
                             const float* __restrict__ pb) {
    int t = blockIdx.x * blockDim.x + threadIdx.x;
    if (t >= 64 * 10) return;
    int o = t % 10;
    out[t] = b0[o] + pb[o] + pb[10 + o] + pb[20 + o] + pb[30 + o];
}

// ---------------- edge scatter: pooled[dst] += h[src] ----------------------
__global__ void k_scatter(const float* __restrict__ h, const int* __restrict__ src,
                          const int* __restrict__ dst, float* pooled,
                          int E, int Dq /* Din/4 */) {
    int idx = blockIdx.x * blockDim.x + threadIdx.x;
    if (idx >= E * Dq) return;
    int e = idx / Dq, f = idx % Dq;
    int s = src[e], d = dst[e];
    const float4 v = *(const float4*)(h + (size_t)s * (Dq * 4) + f * 4);
    float* pd = pooled + (size_t)d * (Dq * 4) + f * 4;
    gatomic_add(pd + 0, v.x);
    gatomic_add(pd + 1, v.y);
    gatomic_add(pd + 2, v.z);
    gatomic_add(pd + 3, v.w);
}

// ---------------- graph sum-pool: gp[gid[n]] += h[n] -----------------------
__global__ void k_pool(const float* __restrict__ h, const int* __restrict__ gid,
                       float* gp, int N, int Dq) {
    int idx = blockIdx.x * blockDim.x + threadIdx.x;
    if (idx >= N * Dq) return;
    int n = idx / Dq, f = idx % Dq;
    int g = gid[n];
    const float4 v = *(const float4*)(h + (size_t)n * (Dq * 4) + f * 4);
    float* pd = gp + (size_t)g * (Dq * 4) + f * 4;
    gatomic_add(pd + 0, v.x);
    gatomic_add(pd + 1, v.y);
    gatomic_add(pd + 2, v.z);
    gatomic_add(pd + 3, v.w);
}

// ---------------- f32 -> bf16 with zero-padding of tail rows ---------------
__global__ void k_cvt_pad(const float* __restrict__ in, __bf16* out,
                          int nrows, int total, int D) {
    int i = blockIdx.x * blockDim.x + threadIdx.x;
    if (i >= total) return;
    int row = i / D;
    out[i] = (row < nrows) ? (__bf16)in[i] : (__bf16)0.0f;
}

// ---------------- weight transpose+convert: out[n*K+k] = w[k*256+n] --------
__global__ void k_wt(const float* __restrict__ w, __bf16* out, int K) {
    int i = blockIdx.x * blockDim.x + threadIdx.x;
    if (i >= 256 * K) return;
    int n = i / K, k = i % K;
    out[i] = (__bf16)w[(size_t)k * 256 + n];
}

// ---------------- WMMA GEMM: Y[MPAD x 256] = A[MPAD x K] @ W[K x 256] + b --
// block = 256 threads (8 waves), grid (MPAD/64, 2).
// The 64 x K A tile is staged ONCE per block into LDS with async
// global->LDS copies (ASYNCcnt), then all 8 waves feed their WMMA A
// fragments from LDS while B streams from the L2-hot transposed weights.
__global__ __launch_bounds__(256) void k_gemm(const __bf16* __restrict__ A,
                                              const __bf16* __restrict__ WT,
                                              const float* __restrict__ bias,
                                              float* __restrict__ Y, int K) {
    __shared__ __bf16 Atile[64 * (HID + 8)];     // padded rows: stride K+8
    const int S = K + 8;                          // bank-conflict-free stride
    const int t = threadIdx.x;
    const int mbase = blockIdx.x * 64;

    // ---- async stage of A tile (64 rows x K) into LDS ----
    const int chunksPerRow = K >> 3;              // 16B chunks per row
    for (int c = t; c < 64 * chunksPerRow; c += 256) {
        int row = c / chunksPerRow;
        int cc  = c - row * chunksPerRow;
        const __bf16* g = A + (size_t)(mbase + row) * K + cc * 8;
        unsigned int lds = (unsigned int)(uintptr_t)&Atile[row * S + cc * 8];
        asm volatile("global_load_async_to_lds_b128 %0, %1, off"
                     :: "v"(lds), "v"((unsigned long long)(uintptr_t)g)
                     : "memory");
    }
    asm volatile("s_wait_asynccnt 0x0" ::: "memory");
    __syncthreads();

    const int wave = t >> 5;
    const int lane = t & 31;
    const int half = lane >> 4;
    const int r    = lane & 15;
    const int ncol = blockIdx.y * 128 + wave * 16;

    v8f c0 = {}, c1 = {}, c2 = {}, c3 = {};
    const __bf16* bp = WT + (size_t)(ncol + r) * K;    // column ncol+r of W

    for (int k = 0; k < K; k += 32) {
        Frag b;
        // B operand: lane group holds K chunks {half*8..+7, 16+half*8..+7}
        b.q[0] = *(const uint4*)(bp + k + half * 8);
        b.q[1] = *(const uint4*)(bp + k + half * 8 + 16);
        // A operand from LDS: lane group holds K chunks {half*8..+7, 16+half*8..+7}
        const __bf16* a = &Atile[r * S + k + half * 8];
        Frag a0, a1, a2, a3;
        a0.q[0] = *(const uint4*)(a);
        a0.q[1] = *(const uint4*)(a + 16);
        a1.q[0] = *(const uint4*)(a + 16 * S);
        a1.q[1] = *(const uint4*)(a + 16 * S + 16);
        a2.q[0] = *(const uint4*)(a + 32 * S);
        a2.q[1] = *(const uint4*)(a + 32 * S + 16);
        a3.q[0] = *(const uint4*)(a + 48 * S);
        a3.q[1] = *(const uint4*)(a + 48 * S + 16);
        c0 = __builtin_amdgcn_wmma_f32_16x16x32_bf16(false, a0.v, false, b.v, (short)0, c0, false, false);
        c1 = __builtin_amdgcn_wmma_f32_16x16x32_bf16(false, a1.v, false, b.v, (short)0, c1, false, false);
        c2 = __builtin_amdgcn_wmma_f32_16x16x32_bf16(false, a2.v, false, b.v, (short)0, c2, false, false);
        c3 = __builtin_amdgcn_wmma_f32_16x16x32_bf16(false, a3.v, false, b.v, (short)0, c3, false, false);
    }
    const float bv = bias[ncol + r];
    float* y0 = Y + (size_t)(mbase + half * 8) * HID + ncol + r;
#pragma unroll
    for (int j = 0; j < 8; ++j) {
        y0[(size_t)j * HID]        = c0[j] + bv;   // rows mbase +  0..15
        y0[(size_t)(j + 16) * HID] = c1[j] + bv;   // rows mbase + 16..31
        y0[(size_t)(j + 32) * HID] = c2[j] + bv;   // rows mbase + 32..47
        y0[(size_t)(j + 48) * HID] = c3[j] + bv;   // rows mbase + 48..63
    }
}

// ---------------- per-column sum / sumsq for BatchNorm ---------------------
__global__ void k_colstats(const float* __restrict__ Y, float* stats,
                           int nrows, int rpb) {
    int c  = threadIdx.x;               // 256 threads = 256 columns
    int r0 = blockIdx.x * rpb;
    int r1 = r0 + rpb; if (r1 > nrows) r1 = nrows;
    float s = 0.f, ss = 0.f;
    for (int r = r0; r < r1; ++r) {
        float v = Y[(size_t)r * HID + c];
        s += v; ss += v * v;
    }
    gatomic_add(&stats[c], s);
    gatomic_add(&stats[HID + c], ss);
}

// ---------------- BN(train stats) + ReLU -> bf16 (padded rows -> 0) --------
__global__ void k_bn_bf16(const float* __restrict__ Y, const float* __restrict__ stats,
                          const float* __restrict__ g, const float* __restrict__ b,
                          __bf16* out, int nrows, int total, float invN) {
    int i = blockIdx.x * blockDim.x + threadIdx.x;
    if (i >= total) return;
    int row = i >> 8, c = i & 255;
    if (row >= nrows) { out[i] = (__bf16)0.0f; return; }
    float m  = stats[c] * invN;
    float vr = stats[HID + c] * invN - m * m;
    float t  = (Y[i] - m) * rsqrtf(vr + 1e-5f) * g[c] + b[c];
    out[i] = (__bf16)(t > 0.f ? t : 0.f);
}

// ---------------- BN + ReLU -> f32 (next-layer h) --------------------------
__global__ void k_bn_f32(const float* __restrict__ Y, const float* __restrict__ stats,
                         const float* __restrict__ g, const float* __restrict__ b,
                         float* out, int total, float invN) {
    int i = blockIdx.x * blockDim.x + threadIdx.x;
    if (i >= total) return;
    int c = i & 255;
    float m  = stats[c] * invN;
    float vr = stats[HID + c] * invN - m * m;
    float t  = (Y[i] - m) * rsqrtf(vr + 1e-5f) * g[c] + b[c];
    out[i] = t > 0.f ? t : 0.f;
}

// ---------------- tiny prediction head: score += gp @ W --------------------
__global__ void k_head(const float* __restrict__ gp, const float* __restrict__ W,
                       float* score, int D) {
    int t = blockIdx.x * blockDim.x + threadIdx.x;
    if (t >= 64 * 10) return;
    int g = t / 10, o = t % 10;
    float s = 0.f;
    for (int d = 0; d < D; ++d) s += gp[(size_t)g * D + d] * W[(size_t)d * 10 + o];
    score[t] += s;
}

// ===========================================================================
extern "C" void kernel_launch(void* const* d_in, const int* in_sizes, int n_in,
                              void* d_out, int out_size, void* d_ws, size_t ws_size,
                              hipStream_t stream) {
    const float* x        = (const float*)d_in[0];
    const float* mlp0_w1  = (const float*)d_in[1];
    const float* mlp0_b1  = (const float*)d_in[2];
    const float* mlp0_g1  = (const float*)d_in[3];
    const float* mlp0_be1 = (const float*)d_in[4];
    const float* mlp0_w2  = (const float*)d_in[5];
    const float* mlp0_b2  = (const float*)d_in[6];
    const float* mlps_w1  = (const float*)d_in[7];
    const float* mlps_b1  = (const float*)d_in[8];
    const float* mlps_g1  = (const float*)d_in[9];
    const float* mlps_be1 = (const float*)d_in[10];
    const float* mlps_w2  = (const float*)d_in[11];
    const float* mlps_b2  = (const float*)d_in[12];
    const float* bn_g     = (const float*)d_in[13];
    const float* bn_b     = (const float*)d_in[14];
    const float* pred_w0  = (const float*)d_in[15];
    const float* pred_b0  = (const float*)d_in[16];
    const float* pred_w   = (const float*)d_in[17];
    const float* pred_b   = (const float*)d_in[18];
    const int*   eidx     = (const int*)d_in[19];
    const int*   gid      = (const int*)d_in[20];
    float* out = (float*)d_out;

    const int N = in_sizes[20];           // 50000 nodes
    const int E = in_sizes[19] / 2;       // 800000 edges
    const int MPAD = ((N + 63) / 64) * 64;
    const float invN = 1.0f / (float)N;
    const int* src = eidx;
    const int* dst = eidx + E;

    // ---- workspace carve-up (256B aligned chunks) ----
    char* w = (char*)d_ws;
    float*  pooled = (float*)w;  w += (size_t)MPAD * HID * 4;
    float*  ybuf   = (float*)w;  w += (size_t)MPAD * HID * 4;
    float*  hbuf   = (float*)w;  w += (size_t)MPAD * HID * 4;
    __bf16* abf    = (__bf16*)w; w += (size_t)MPAD * HID * 2;
    __bf16* wbf    = (__bf16*)w; w += (size_t)HID * HID * 2;
    float*  stats  = (float*)w;  w += 512 * 4;
    float*  gp     = (float*)w;  w += (size_t)64 * HID * 4;

    const int T = 256;
    auto blk = [](int n, int t) { return (n + t - 1) / t; };

    // score = all biases
    k_init_score<<<blk(640, T), T, 0, stream>>>(out, pred_b0, pred_b);
    // head for hidden_rep[0] = x (D=128)
    k_zero<<<blk(64 * 128, T), T, 0, stream>>>(gp, 64 * 128);
    k_pool<<<blk(N * 32, T), T, 0, stream>>>(x, gid, gp, N, 32);
    k_head<<<blk(640, T), T, 0, stream>>>(gp, pred_w0, out, 128);

    const float* h = x;
    int Din = 128;
    for (int l = 0; l < 4; ++l) {
        const float *w1, *b1, *g1, *be1, *w2, *b2;
        if (l == 0) {
            w1 = mlp0_w1; b1 = mlp0_b1; g1 = mlp0_g1; be1 = mlp0_be1;
            w2 = mlp0_w2; b2 = mlp0_b2;
        } else {
            int j = l - 1;
            w1 = mlps_w1 + (size_t)j * HID * HID; b1 = mlps_b1 + j * HID;
            g1 = mlps_g1 + j * HID;               be1 = mlps_be1 + j * HID;
            w2 = mlps_w2 + (size_t)j * HID * HID; b2 = mlps_b2 + j * HID;
        }
        const int Dq = Din / 4;

        // pooled = h (self loop) + scatter-add over edges
        k_copy4<<<blk(N * Dq, T), T, 0, stream>>>((const float4*)h, (float4*)pooled, N * Dq);
        k_scatter<<<blk(E * Dq, T), T, 0, stream>>>(h, src, dst, pooled, E, Dq);

        // GEMM 1: y = bf16(pooled) @ bf16(w1) + b1
        k_cvt_pad<<<blk(MPAD * Din, T), T, 0, stream>>>(pooled, abf, N, MPAD * Din, Din);
        k_wt<<<blk(256 * Din, T), T, 0, stream>>>(w1, wbf, Din);
        dim3 gg(MPAD / 64, 2);
        k_gemm<<<gg, T, 0, stream>>>(abf, wbf, b1, ybuf, Din);

        // BN + ReLU -> t (bf16)
        k_zero<<<blk(512, T), T, 0, stream>>>(stats, 512);
        k_colstats<<<blk(N, 512), T, 0, stream>>>(ybuf, stats, N, 512);
        k_bn_bf16<<<blk(MPAD * HID, T), T, 0, stream>>>(ybuf, stats, g1, be1, abf, N,
                                                        MPAD * HID, invN);

        // GEMM 2: y = t @ bf16(w2) + b2
        k_wt<<<blk(256 * HID, T), T, 0, stream>>>(w2, wbf, HID);
        k_gemm<<<gg, T, 0, stream>>>(abf, wbf, b2, ybuf, HID);

        // outer BN + ReLU -> h (f32)
        k_zero<<<blk(512, T), T, 0, stream>>>(stats, 512);
        k_colstats<<<blk(N, 512), T, 0, stream>>>(ybuf, stats, N, 512);
        k_bn_f32<<<blk(N * HID, T), T, 0, stream>>>(ybuf, stats, bn_g + l * HID,
                                                    bn_b + l * HID, hbuf, N * HID, invN);
        h = hbuf; Din = HID;

        // head for hidden_rep[l+1]
        k_zero<<<blk(64 * HID, T), T, 0, stream>>>(gp, 64 * HID);
        k_pool<<<blk(N * 64, T), T, 0, stream>>>(hbuf, gid, gp, N, 64);
        k_head<<<blk(640, T), T, 0, stream>>>(gp, pred_w + (size_t)l * HID * 10, out, HID);
    }
}
